// BlockRC1_16140487099013
// MI455X (gfx1250) — matrix operand order repaired and checked by reading, not compiled, not run on hardware
//
#include <hip/hip_runtime.h>
#include <math.h>

typedef _Float16 f16;
typedef _Float16 v16h __attribute__((ext_vector_type(16)));
typedef _Float16 h8 __attribute__((ext_vector_type(8)));
typedef float v8f __attribute__((ext_vector_type(8)));

union AF { v16h v; h8 h[2]; f16 e[16]; };
union CF { v8f v; float e[8]; };

#define T_TOT 65536
#define NB 2
#define LDA 72   // padded leading dim (f16 elems): 144 B = 16B-aligned, 36-bank stride

__device__ inline float gelu_f(float x) {
  return 0.5f * x * (1.0f + erff(x * 0.70710678118654752f));
}

// A-operand: 16x32 f16 tile from row-major LDS (ld elems, rows 16B-aligned).
// lanes 0-15: row M=lane, K=k0+{0..7,16..23}; lanes 16-31: row M=lane-16, K=k0+{8..15,24..31}.
__device__ inline v16h ldsA(const f16* __restrict__ s, int ld, int row0, int k0, int lane) {
  AF a;
  const int r = row0 + (lane & 15);
  const int kb = k0 + ((lane & 16) ? 8 : 0);
  const f16* p = s + r * ld + kb;
  a.h[0] = *(const h8*)(p);
  a.h[1] = *(const h8*)(p + 16);
  return a.v;
}

// B-operand from TRANSPOSED (N-major) LDS: sT[n*ld + k].
// lanes 0-15: col N=lane, K=k0+{0..7,16..23} ; lanes 16-31: col N=lane-16, K=k0+{8..15,24..31}.
__device__ inline v16h ldsBT(const f16* __restrict__ s, int ld, int k0, int col0, int lane) {
  AF b;
  const int n = col0 + (lane & 15);
  const int kb = k0 + ((lane & 16) ? 8 : 0);
  const f16* p = s + n * ld + kb;
  b.h[0] = *(const h8*)(p);
  b.h[1] = *(const h8*)(p + 16);
  return b.v;
}

// C/D layout: reg r, lanes 0-15 -> (M=r, N=lane), lanes 16-31 -> (M=8+r, N=lane-16).

// ---------------- Kernel A: 4 dilated convs + bias + gelu -> tokens (B,T,256) ----------------
__global__ __launch_bounds__(256) void conv_gelu_kernel(
    const float* __restrict__ x, const float* __restrict__ cw,
    const float* __restrict__ cb, float* __restrict__ tokens) {
  const int dils[4] = {1, 2, 3, 4};
  const int pads[4] = {2, 5, 8, 11};
  const int ow = threadIdx.x;
  const int oh = blockIdx.x;
  const int ci = blockIdx.y;
  const int b  = blockIdx.z;

  __shared__ float sW[64 * 49];
  __shared__ float sCb[64];
  for (int idx = threadIdx.x; idx < 64 * 49; idx += 256) sW[idx] = cw[ci * 64 * 49 + idx];
  if (threadIdx.x < 64) sCb[threadIdx.x] = cb[ci * 64 + threadIdx.x];
  __syncthreads();

  const int d = dils[ci], p = pads[ci];
  const float* xb = x + (long)b * 1048576;
  float xv[49];
#pragma unroll
  for (int kh = 0; kh < 7; ++kh) {
    const int ih = oh * 4 - p + kh * d;
#pragma unroll
    for (int kw = 0; kw < 7; ++kw) {
      const int iw = ow * 4 - p + kw * d;
      xv[kh * 7 + kw] = (ih >= 0 && ih < 1024 && iw >= 0 && iw < 1024) ? xb[ih * 1024 + iw] : 0.0f;
    }
  }
  const long t = (long)oh * 256 + ow;
  float* outp = tokens + ((long)b * T_TOT + t) * 256 + ci * 64;
  for (int oc = 0; oc < 64; ++oc) {
    float a = sCb[oc];
    const float* wp = sW + oc * 49;
#pragma unroll
    for (int j = 0; j < 49; ++j) a += wp[j] * xv[j];
    outp[oc] = gelu_f(a);
  }
}

// ---------------- Kernel B: per-token mean / rstd (wave per token) ----------------
__global__ __launch_bounds__(256) void ln_stats_kernel(
    const float* __restrict__ tokens, float* __restrict__ mr) {
  const int lane = threadIdx.x & 31;
  const int wv = threadIdx.x >> 5;
  const long token = (long)blockIdx.x * 8 + wv;
  const float4* row = (const float4*)(tokens + token * 256) + lane * 2;
  float4 f0 = row[0], f1 = row[1];
  float s = f0.x + f0.y + f0.z + f0.w + f1.x + f1.y + f1.z + f1.w;
  float ss = f0.x * f0.x + f0.y * f0.y + f0.z * f0.z + f0.w * f0.w +
             f1.x * f1.x + f1.y * f1.y + f1.z * f1.z + f1.w * f1.w;
#pragma unroll
  for (int off = 16; off; off >>= 1) {
    s += __shfl_xor(s, off, 32);
    ss += __shfl_xor(ss, off, 32);
  }
  if (lane == 0) {
    float mu = s * (1.0f / 256.0f);
    float var = ss * (1.0f / 256.0f) - mu * mu;
    mr[token * 2 + 0] = mu;
    mr[token * 2 + 1] = rsqrtf(var + 1e-5f);
  }
}

// ---------------- Kernel C: LN + kqv GEMM (WMMA) + performer stage-1 ----------------
// Block: 128 tokens, 256 threads (8 waves). Main GEMM: 128x256 @ 256x192, f16 WMMA.
__global__ __launch_bounds__(256) void kqv_performer_kernel(
    const float* __restrict__ tokens, const float* __restrict__ mr,
    const float* __restrict__ n1g, const float* __restrict__ n1b,
    const float* __restrict__ kqv_w, const float* __restrict__ kqv_b,
    const float* __restrict__ wperf,
    float* __restrict__ vbuf, float* __restrict__ qpbuf, float* __restrict__ part) {
  __shared__ __align__(16) unsigned char smem[57344];
  const int tid = threadIdx.x, lane = tid & 31, wv = tid >> 5;
  const int b = blockIdx.y, blk = blockIdx.x;
  const int t0 = blk * 128;
  const long base_tok = ((long)b * T_TOT + t0) * 256;

  // phase-1 aliases
  f16* sA   = (f16*)smem;                 // 128 x LDA f16  (18432 B)
  f16* sWt  = (f16*)(smem + 18432);       // 192 x LDA f16, N-major transposed (27648 B)
  float* sMu = (float*)(smem + 46080);    // 128
  float* sRs = (float*)(smem + 46592);    // 128
  float* sG  = (float*)(smem + 47104);    // 256
  float* sBt = (float*)(smem + 48128);    // 256 (ends 49152)
  // phase-2 aliases
  f16* sVT  = (f16*)smem;                 // v^T 64x128 f16 (16384 B)
  f16* sKQ  = (f16*)(smem + 16384);       // k or q 128 x LDA f16 (18432 B)
  float* sKP = (float*)(smem + 34816);    // kp 128x32 f32 (16384 B)
  f16* sWPt = (f16*)(smem + 51200);       // w_perf 32 x LDA f16, N-major (4608 B)
  float* sXD = (float*)(smem + 55808);    // 128 f32

  if (tid < 128) {
    sMu[tid] = mr[((long)b * T_TOT + t0 + tid) * 2 + 0];
    sRs[tid] = mr[((long)b * T_TOT + t0 + tid) * 2 + 1];
  }
  sG[tid] = n1g[tid];
  sBt[tid] = n1b[tid];

  CF acc[12];
#pragma unroll
  for (int tn = 0; tn < 12; ++tn)
#pragma unroll
    for (int j = 0; j < 8; ++j) acc[tn].e[j] = 0.0f;

  for (int kk = 0; kk < 256; kk += 64) {
    __syncthreads();
    // A chunk: LN(tokens) -> f16, 8 elems per thread-iter via float4 + ds_store_b128
    for (int g = tid; g < 1024; g += 256) {
      const int t = g >> 3, j0 = (g & 7) * 8;
      const float4* pf = (const float4*)(tokens + base_tok + (long)t * 256 + kk + j0);
      const float4 f0 = pf[0], f1 = pf[1];
      const float mu = sMu[t], rs = sRs[t];
      const int c = kk + j0;
      h8 hv;
      hv[0] = (f16)((f0.x - mu) * rs * sG[c + 0] + sBt[c + 0]);
      hv[1] = (f16)((f0.y - mu) * rs * sG[c + 1] + sBt[c + 1]);
      hv[2] = (f16)((f0.z - mu) * rs * sG[c + 2] + sBt[c + 2]);
      hv[3] = (f16)((f0.w - mu) * rs * sG[c + 3] + sBt[c + 3]);
      hv[4] = (f16)((f1.x - mu) * rs * sG[c + 4] + sBt[c + 4]);
      hv[5] = (f16)((f1.y - mu) * rs * sG[c + 5] + sBt[c + 5]);
      hv[6] = (f16)((f1.z - mu) * rs * sG[c + 6] + sBt[c + 6]);
      hv[7] = (f16)((f1.w - mu) * rs * sG[c + 7] + sBt[c + 7]);
      *(h8*)(sA + t * LDA + j0) = hv;
    }
    // B chunk: kqv_w 64x192 -> transposed N-major (coalesced global reads)
    for (int i = tid; i < 12288; i += 256) {
      const int kr = i / 192, n = i - kr * 192;
      sWt[n * LDA + kr] = (f16)kqv_w[(long)(kk + kr) * 192 + n];
    }
    if (kk + 64 < 256)
      __builtin_prefetch(tokens + base_tok + kk + 64 + (long)(tid >> 6) * 256, 0, 1);
    __syncthreads();
#pragma unroll
    for (int k0 = 0; k0 < 64; k0 += 32) {
      v16h a = ldsA(sA, LDA, wv * 16, k0, lane);
      // software-pipelined B fragments: issue next loads before current WMMA
      v16h bcur = ldsBT(sWt, LDA, k0, 0, lane);
#pragma unroll
      for (int tn = 0; tn < 12; ++tn) {
        v16h bnxt = bcur;
        if (tn < 11) bnxt = ldsBT(sWt, LDA, k0, (tn + 1) * 16, lane);
        acc[tn].v = __builtin_amdgcn_wmma_f32_16x16x32_f16(
            false, a, false, bcur, (short)0, acc[tn].v, false, false);
        bcur = bnxt;
      }
    }
  }
  __syncthreads();

  // ---- epilogue: extract k (cols 0..63), v (cols 128..191) ----
  const int cn = lane & 15;
  const int rb = (lane & 16) ? 8 : 0;
#pragma unroll
  for (int tn = 0; tn < 4; ++tn) {
#pragma unroll
    for (int r = 0; r < 8; ++r) {
      const int t = wv * 16 + rb + r;
      const int n = tn * 16 + cn;
      sKQ[t * LDA + n] = (f16)(acc[tn].e[r] + kqv_b[n]);
    }
  }
#pragma unroll
  for (int tn = 8; tn < 12; ++tn) {
#pragma unroll
    for (int r = 0; r < 8; ++r) {
      const int t = wv * 16 + rb + r;
      const int n = tn * 16 + cn;
      const float vvv = acc[tn].e[r] + kqv_b[n];
      const int nv = n - 128;
      vbuf[((long)b * T_TOT + t0 + t) * 64 + nv] = vvv;
      sVT[nv * 128 + t] = (f16)vvv;
    }
  }
  // w_perf (32x64, row-major) -> N-major LDS (identity copy with pad)
  for (int i = tid; i < 2048; i += 256)
    sWPt[(i >> 6) * LDA + (i & 63)] = (f16)wperf[i];
  __syncthreads();

  // xd for k
  if (tid < 128) {
    float s = 0.f;
    for (int j = 0; j < 64; ++j) { float kv = (float)sKQ[tid * LDA + j]; s += kv * kv; }
    sXD[tid] = 0.5f * s;
  }
  __syncthreads();

  // wtx_k = k @ w_perf^T (128x64 @ 64x32) via WMMA; kp = exp(wtx - xd)/sqrt(32) (f32 in LDS)
  CF w0[2];
#pragma unroll
  for (int tn = 0; tn < 2; ++tn)
#pragma unroll
    for (int j = 0; j < 8; ++j) w0[tn].e[j] = 0.0f;
  {
    v16h a0 = ldsA(sKQ, LDA, wv * 16, 0, lane);
    v16h b0 = ldsBT(sWPt, LDA, 0, 0, lane);
    v16h b1 = ldsBT(sWPt, LDA, 0, 16, lane);
    w0[0].v = __builtin_amdgcn_wmma_f32_16x16x32_f16(false, a0, false, b0, (short)0, w0[0].v, false, false);
    v16h a1 = ldsA(sKQ, LDA, wv * 16, 32, lane);
    v16h b2 = ldsBT(sWPt, LDA, 32, 0, lane);
    w0[1].v = __builtin_amdgcn_wmma_f32_16x16x32_f16(false, a0, false, b1, (short)0, w0[1].v, false, false);
    v16h b3 = ldsBT(sWPt, LDA, 32, 16, lane);
    w0[0].v = __builtin_amdgcn_wmma_f32_16x16x32_f16(false, a1, false, b2, (short)0, w0[0].v, false, false);
    w0[1].v = __builtin_amdgcn_wmma_f32_16x16x32_f16(false, a1, false, b3, (short)0, w0[1].v, false, false);
  }
#pragma unroll
  for (int tn = 0; tn < 2; ++tn)
#pragma unroll
    for (int r = 0; r < 8; ++r) {
      const int t = wv * 16 + rb + r;
      const int m = tn * 16 + cn;
      sKP[t * 32 + m] = expf(w0[tn].e[r] - sXD[t]) * 0.17677669529663687f;
    }
  __syncthreads();

  // deterministic per-block partials: kptv[n][m] = sum_t v[t][n]*kp[t][m]; kpsum[m]
  float* pout = part + ((long)(b * 512 + blk)) * 2080;
  for (int e = tid; e < 2048; e += 256) {
    const int n = e >> 5, m = e & 31;
    float s = 0.f;
    for (int t = 0; t < 128; ++t) s += (float)sVT[n * 128 + t] * sKP[t * 32 + m];
    pout[e] = s;
  }
  if (tid < 32) {
    float s = 0.f;
    for (int t = 0; t < 128; ++t) s += sKP[t * 32 + tid];
    pout[2048 + tid] = s;
  }
  // overwrite sKQ with q (cols 64..127) — safe: all waves past wtx_k reads
#pragma unroll
  for (int tn = 4; tn < 8; ++tn) {
#pragma unroll
    for (int r = 0; r < 8; ++r) {
      const int t = wv * 16 + rb + r;
      const int n = tn * 16 + cn;
      sKQ[t * LDA + (n - 64)] = (f16)(acc[tn].e[r] + kqv_b[n]);
    }
  }
  __syncthreads();

  if (tid < 128) {
    float s = 0.f;
    for (int j = 0; j < 64; ++j) { float qv = (float)sKQ[tid * LDA + j]; s += qv * qv; }
    sXD[tid] = 0.5f * s;
  }
  __syncthreads();

#pragma unroll
  for (int tn = 0; tn < 2; ++tn)
#pragma unroll
    for (int j = 0; j < 8; ++j) w0[tn].e[j] = 0.0f;
  {
    v16h a0 = ldsA(sKQ, LDA, wv * 16, 0, lane);
    v16h b0 = ldsBT(sWPt, LDA, 0, 0, lane);
    v16h b1 = ldsBT(sWPt, LDA, 0, 16, lane);
    w0[0].v = __builtin_amdgcn_wmma_f32_16x16x32_f16(false, a0, false, b0, (short)0, w0[0].v, false, false);
    v16h a1 = ldsA(sKQ, LDA, wv * 16, 32, lane);
    v16h b2 = ldsBT(sWPt, LDA, 32, 0, lane);
    w0[1].v = __builtin_amdgcn_wmma_f32_16x16x32_f16(false, a0, false, b1, (short)0, w0[1].v, false, false);
    v16h b3 = ldsBT(sWPt, LDA, 32, 16, lane);
    w0[0].v = __builtin_amdgcn_wmma_f32_16x16x32_f16(false, a1, false, b2, (short)0, w0[0].v, false, false);
    w0[1].v = __builtin_amdgcn_wmma_f32_16x16x32_f16(false, a1, false, b3, (short)0, w0[1].v, false, false);
  }
#pragma unroll
  for (int tn = 0; tn < 2; ++tn)
#pragma unroll
    for (int r = 0; r < 8; ++r) {
      const int t = wv * 16 + rb + r;
      const int m = tn * 16 + cn;
      qpbuf[((long)b * T_TOT + t0 + t) * 32 + m] =
          expf(w0[tn].e[r] - sXD[t]) * 0.17677669529663687f;
    }
}

// ---------------- Kernel D: deterministic reduction of partials ----------------
__global__ __launch_bounds__(256) void reduce_partials_kernel(
    const float* __restrict__ part, float* __restrict__ kptv_g, float* __restrict__ kpsum_g) {
  const int b = blockIdx.x;
  for (int e = threadIdx.x; e < 2080; e += 256) {
    float s = 0.f;
    const float* p = part + ((long)b * 512) * 2080 + e;
    for (int blk = 0; blk < 512; ++blk) s += p[(long)blk * 2080];
    if (e < 2048) kptv_g[b * 2048 + e] = s;
    else kpsum_g[b * 32 + (e - 2048)] = s;
  }
}

// 64x64 GEMM helper: A row-major 64xLDA f16, B transposed N-major 64xLDA f16.
// Software-pipelined: loads interleaved between WMMAs.
__device__ inline void gemm64(const f16* __restrict__ sA, const f16* __restrict__ sB,
                              int tm, int tn0, int lane, CF& c0, CF& c1) {
#pragma unroll
  for (int j = 0; j < 8; ++j) { c0.e[j] = 0.0f; c1.e[j] = 0.0f; }
  v16h a0 = ldsA(sA, LDA, tm * 16, 0, lane);
  v16h b0 = ldsBT(sB, LDA, 0, tn0 * 16, lane);
  v16h b1 = ldsBT(sB, LDA, 0, tn0 * 16 + 16, lane);
  c0.v = __builtin_amdgcn_wmma_f32_16x16x32_f16(false, a0, false, b0, (short)0, c0.v, false, false);
  v16h a1 = ldsA(sA, LDA, tm * 16, 32, lane);
  v16h b2 = ldsBT(sB, LDA, 32, tn0 * 16, lane);
  c1.v = __builtin_amdgcn_wmma_f32_16x16x32_f16(false, a0, false, b1, (short)0, c1.v, false, false);
  v16h b3 = ldsBT(sB, LDA, 32, tn0 * 16 + 16, lane);
  c0.v = __builtin_amdgcn_wmma_f32_16x16x32_f16(false, a1, false, b2, (short)0, c0.v, false, false);
  c1.v = __builtin_amdgcn_wmma_f32_16x16x32_f16(false, a1, false, b3, (short)0, c1.v, false, false);
}

// ---------------- Kernel E: attention output + proj + LN + MLP (WMMA) ----------------
// Block: 64 tokens, 256 threads (8 waves); wave w -> tile row w>>1, tile cols 2*(w&1)+{0,1}
__global__ __launch_bounds__(256) void attn_mlp_kernel(
    const float* __restrict__ qpbuf, const float* __restrict__ vbuf,
    const float* __restrict__ kptv_g, const float* __restrict__ kpsum_g,
    const float* __restrict__ proj_w, const float* __restrict__ proj_b,
    const float* __restrict__ n2g, const float* __restrict__ n2b,
    const float* __restrict__ mlp1_w, const float* __restrict__ mlp1_b,
    const float* __restrict__ mlp2_w, const float* __restrict__ mlp2_b,
    float* __restrict__ out) {
  __shared__ float sQp[64 * 32];
  __shared__ float sKtv[64 * 32];
  __shared__ float sKs[32];
  __shared__ float sD[64];
  __shared__ __align__(16) f16 sA[64 * LDA];
  __shared__ __align__(16) f16 sB[64 * LDA];   // weights, N-major transposed
  __shared__ float sY[64 * 64];

  const int tid = threadIdx.x, lane = tid & 31, wv = tid >> 5;
  const int b = blockIdx.y;
  const int t0 = blockIdx.x * 64;
  const int tm = wv >> 1, tn0 = (wv & 1) * 2;
  const int cn = lane & 15;
  const int rb = (lane & 16) ? 8 : 0;

  for (int i = tid; i < 2048; i += 256) {
    const int t = i >> 5, m = i & 31;
    sQp[i] = qpbuf[((long)b * T_TOT + t0 + t) * 32 + m];
    sKtv[i] = kptv_g[b * 2048 + i];
  }
  if (tid < 32) sKs[tid] = kpsum_g[b * 32 + tid];
  for (int i = tid; i < 4096; i += 256)
    sB[(i & 63) * LDA + (i >> 6)] = (f16)proj_w[i];
  __syncthreads();

  if (tid < 64) {
    float s = 0.f;
    for (int m = 0; m < 32; ++m) s += sQp[tid * 32 + m] * sKs[m];
    sD[tid] = s;
  }
  __syncthreads();

  // y_att[t][n] = (sum_m qp[t][m]*kptv[n][m]) / (D+1e-8)  (scalar f32: kp/qp exceed f16 range)
  for (int i = tid; i < 4096; i += 256) {
    const int t = i >> 6, n = i & 63;
    float s = 0.f;
    for (int m = 0; m < 32; ++m) s += sQp[t * 32 + m] * sKtv[n * 32 + m];
    sA[t * LDA + n] = (f16)(s / (sD[t] + 1e-8f));
  }
  __syncthreads();

  CF c0, c1;
  // proj: y = v + y_att @ proj_w + proj_b
  gemm64(sA, sB, tm, tn0, lane, c0, c1);
#pragma unroll
  for (int h = 0; h < 2; ++h) {
    CF& cc = h ? c1 : c0;
    const int n = (tn0 + h) * 16 + cn;
#pragma unroll
    for (int r = 0; r < 8; ++r) {
      const int t = tm * 16 + rb + r;
      sY[t * 64 + n] = vbuf[((long)b * T_TOT + t0 + t) * 64 + n] + cc.e[r] + proj_b[n];
    }
  }
  __syncthreads();

  // LN(64) -> sA f16 ; load mlp1_w (transposed)
  if (tid < 64) {
    float s = 0.f, ss = 0.f;
    for (int n = 0; n < 64; ++n) { float yv = sY[tid * 64 + n]; s += yv; ss += yv * yv; }
    const float mu = s * (1.0f / 64.0f);
    const float rs = rsqrtf(ss * (1.0f / 64.0f) - mu * mu + 1e-5f);
    for (int n = 0; n < 64; ++n)
      sA[tid * LDA + n] = (f16)(((sY[tid * 64 + n] - mu) * rs) * n2g[n] + n2b[n]);
  }
  for (int i = tid; i < 4096; i += 256)
    sB[(i & 63) * LDA + (i >> 6)] = (f16)mlp1_w[i];
  __syncthreads();

  gemm64(sA, sB, tm, tn0, lane, c0, c1);
  __syncthreads();
#pragma unroll
  for (int h = 0; h < 2; ++h) {
    CF& cc = h ? c1 : c0;
    const int n = (tn0 + h) * 16 + cn;
#pragma unroll
    for (int r = 0; r < 8; ++r) {
      const int t = tm * 16 + rb + r;
      sA[t * LDA + n] = (f16)gelu_f(cc.e[r] + mlp1_b[n]);
    }
  }
  for (int i = tid; i < 4096; i += 256)
    sB[(i & 63) * LDA + (i >> 6)] = (f16)mlp2_w[i];
  __syncthreads();

  gemm64(sA, sB, tm, tn0, lane, c0, c1);
#pragma unroll
  for (int h = 0; h < 2; ++h) {
    CF& cc = h ? c1 : c0;
    const int n = (tn0 + h) * 16 + cn;
#pragma unroll
    for (int r = 0; r < 8; ++r) {
      const int t = tm * 16 + rb + r;
      out[((long)b * T_TOT + t0 + t) * 64 + n] = sY[t * 64 + n] + cc.e[r] + mlp2_b[n];
    }
  }
}

extern "C" void kernel_launch(void* const* d_in, const int* in_sizes, int n_in,
                              void* d_out, int out_size, void* d_ws, size_t ws_size,
                              hipStream_t stream) {
  const float* x      = (const float*)d_in[0];
  const float* conv_w = (const float*)d_in[1];
  const float* conv_b = (const float*)d_in[2];
  const float* n1g    = (const float*)d_in[3];
  const float* n1b    = (const float*)d_in[4];
  const float* kqv_w  = (const float*)d_in[5];
  const float* kqv_b  = (const float*)d_in[6];
  const float* proj_w = (const float*)d_in[7];
  const float* proj_b = (const float*)d_in[8];
  const float* n2g    = (const float*)d_in[9];
  const float* n2b    = (const float*)d_in[10];
  const float* mlp1_w = (const float*)d_in[11];
  const float* mlp1_b = (const float*)d_in[12];
  const float* mlp2_w = (const float*)d_in[13];
  const float* mlp2_b = (const float*)d_in[14];
  const float* wperf  = (const float*)d_in[15];

  char* ws = (char*)d_ws;
  size_t off = 0;
  float* tokens = (float*)(ws + off); off += (size_t)NB * T_TOT * 256 * 4;  // 134 MB
  float* mr     = (float*)(ws + off); off += (size_t)NB * T_TOT * 2 * 4;    // 1 MB
  float* vbuf   = (float*)(ws + off); off += (size_t)NB * T_TOT * 64 * 4;   // 33.5 MB
  float* qpbuf  = (float*)(ws + off); off += (size_t)NB * T_TOT * 32 * 4;   // 16.8 MB
  float* part   = (float*)(ws + off); off += (size_t)NB * 512 * 2080 * 4;   // 8.5 MB
  float* kptv_g = (float*)(ws + off); off += (size_t)NB * 2048 * 4;
  float* kpsum_g= (float*)(ws + off); off += (size_t)NB * 32 * 4;
  (void)ws_size; (void)in_sizes; (void)n_in; (void)out_size;

  conv_gelu_kernel<<<dim3(256, 4, NB), 256, 0, stream>>>(x, conv_w, conv_b, tokens);
  ln_stats_kernel<<<dim3(NB * T_TOT / 8), 256, 0, stream>>>(tokens, mr);
  kqv_performer_kernel<<<dim3(512, NB), 256, 0, stream>>>(
      tokens, mr, n1g, n1b, kqv_w, kqv_b, wperf, vbuf, qpbuf, part);
  reduce_partials_kernel<<<dim3(NB), 256, 0, stream>>>(part, kptv_g, kpsum_g);
  attn_mlp_kernel<<<dim3(1024, NB), 256, 0, stream>>>(
      qpbuf, vbuf, kptv_g, kpsum_g, proj_w, proj_b, n2g, n2b,
      mlp1_w, mlp1_b, mlp2_w, mlp2_b, (float*)d_out);
}
